// E2AttentionNodeFMM_77704548320004
// MI455X (gfx1250) — matrix operand order repaired and checked by reading, not compiled, not run on hardware
//
#include <hip/hip_runtime.h>
#include <cmath>
#include <complex>
#include <cstring>
#include <algorithm>

// ---------------- problem constants ----------------
#define NN    2048   // nodes
#define HH    8      // heads
#define NCD   9      // (LMAX+1)^2
#define UUQ   25     // quadrature points
#define KKB   8      // kappa bands
#define CLD   144    // CV*NC
#define RWS   1600   // 2(trig)*U*DQK combined contraction rows

typedef __attribute__((ext_vector_type(16))) _Float16 v16h;
typedef __attribute__((ext_vector_type(8)))  _Float16 v8h;
typedef __attribute__((ext_vector_type(8)))  float    v8f;
typedef unsigned int u32x4 __attribute__((ext_vector_type(4)));
typedef int          i32x4 __attribute__((ext_vector_type(4)));
typedef int          i32x8 __attribute__((ext_vector_type(8)));

// Tensor Data Mover availability (device pass only; fallback = cooperative copy)
#if defined(__HIP_DEVICE_COMPILE__)
#if __has_builtin(__builtin_amdgcn_tensor_load_to_lds) && __has_builtin(__builtin_amdgcn_s_wait_tensorcnt)
#define HAVE_TDM 1
#endif
#endif

#ifdef HAVE_TDM
#if __clang_major__ >= 23
#define TDM_LOAD(g0, g1, z4) do { i32x8 _z8 = {0,0,0,0,0,0,0,0}; \
  __builtin_amdgcn_tensor_load_to_lds((g0), (g1), (z4), (z4), _z8, 0); } while (0)
#else
#define TDM_LOAD(g0, g1, z4) __builtin_amdgcn_tensor_load_to_lds((g0), (g1), (z4), (z4), 0)
#endif
#endif

#define WMMA_F16(a, b, c) \
  __builtin_amdgcn_wmma_f32_16x16x32_f16(false, (a), false, (b), (short)0, (c), false, false)

// ---- fragment loaders (ISA 7.12.2, wave32). All operands stored K-contiguous. ----
// A (16x32): lane -> row l&15; K halves at (l>>4)*8+{0..7} and 16+(l>>4)*8+{0..7}
__device__ __forceinline__ v16h frag_a_rowmajor(const _Float16* p, int stride) {
  const int l = threadIdx.x & 31;
  const _Float16* q = p + (size_t)(l & 15) * stride + ((l >> 4) * 8);
  v8h lo = *(const v8h*)(q);
  v8h hi = *(const v8h*)(q + 16);
  v16h r;
#pragma unroll
  for (int i = 0; i < 8; ++i) { r[i] = lo[i]; r[i + 8] = hi[i]; }
  return r;
}
// B (32x16): lane -> col l&15 (= row of B^T); K halves at (l>>4)*8+{0..7} and 16+(l>>4)*8+{0..7}
__device__ __forceinline__ v16h frag_b_kcontig(const _Float16* p, int stride) {
  const int l = threadIdx.x & 31;
  const _Float16* q = p + (size_t)(l & 15) * stride + ((l >> 4) * 8);
  v8h lo = *(const v8h*)(q);
  v8h hi = *(const v8h*)(q + 16);
  v16h r;
#pragma unroll
  for (int i = 0; i < 8; ++i) { r[i] = lo[i]; r[i + 8] = hi[i]; }
  return r;
}
__device__ __forceinline__ v16h pack_ab(v8h lo, v8h hi) {
  v16h r;
#pragma unroll
  for (int i = 0; i < 8; ++i) { r[i] = lo[i]; r[i + 8] = hi[i]; }
  return r;
}

// ---------------- prep kernels ----------------
// transpose-cast: src (R x C) f32 -> dst (C x R) f16
__global__ void t16_kernel(const float* __restrict__ src, _Float16* __restrict__ dst, int R, int C) {
  int idx = blockIdx.x * 256 + threadIdx.x;
  if (idx >= R * C) return;
  int c = idx / R, r = idx % R;
  dst[idx] = (_Float16)src[(size_t)r * C + c];
}

// WqkT[j][i*256+c]: j<256 -> Wq[i][c][j], else Wk[i][c][j-256]   (512 x 768, K-contig)
__global__ void pack_wqkT_kernel(const float* __restrict__ Wq, const float* __restrict__ Wk,
                                 _Float16* __restrict__ WqkT) {
  int idx = blockIdx.x * 256 + threadIdx.x;  // 512*768
  int j = idx / 768, rr = idx % 768;
  int i = rr >> 8, c = rr & 255;
  float v = (j < 256) ? Wq[((size_t)i * 256 + c) * 256 + j]
                      : Wk[((size_t)i * 256 + c) * 256 + (j - 256)];
  WqkT[idx] = (_Float16)v;
}

// B2[kk][u][h][l][m] = sum_e YW[e,u] * a[ldeg(e),h,kk] * RRED[e,l,m]
__global__ void b2_kernel(const float* __restrict__ YW, const float* __restrict__ RRED,
                          const float* __restrict__ a, float* __restrict__ B2) {
  int idx = blockIdx.x * 256 + threadIdx.x;
  if (idx >= KKB * UUQ * HH * 81) return;
  int m  = idx % 9;
  int lq = (idx / 9) % 9;
  int h  = (idx / 81) % HH;
  int u  = (idx / (81 * HH)) % UUQ;
  int kk = idx / (81 * HH * UUQ);
  const int ldeg[9] = {0, 1, 1, 1, 2, 2, 2, 2, 2};
  float s = 0.f;
#pragma unroll
  for (int e = 0; e < 9; ++e)
    s += YW[e * UUQ + u] * a[(ldeg[e] * HH + h) * KKB + kk] * RRED[(e * 9 + lq) * 9 + m];
  B2[idx] = s;
}

// S16 (invariants), feat16 (f16 copy), cs16T[(kk,t,u)][n] (f16 phases)
__global__ void prep_kernel(const float* __restrict__ pos, const float* __restrict__ feat,
                            const float* __restrict__ kappa, const float* __restrict__ DIRS,
                            _Float16* __restrict__ S16, _Float16* __restrict__ feat16,
                            _Float16* __restrict__ cs16T) {
  int n = blockIdx.x;
  int c = threadIdx.x;  // 256
  float x[9];
#pragma unroll
  for (int m = 0; m < 9; ++m) {
    x[m] = feat[((size_t)n * 9 + m) * 256 + c];
    feat16[((size_t)n * 9 + m) * 256 + c] = (_Float16)x[m];
  }
  float s1 = sqrtf(x[1] * x[1] + x[2] * x[2] + x[3] * x[3] + 1e-8f);
  float s2 = sqrtf(x[4] * x[4] + x[5] * x[5] + x[6] * x[6] + x[7] * x[7] + x[8] * x[8] + 1e-8f);
  S16[(size_t)n * 768 + c]       = (_Float16)x[0];
  S16[(size_t)n * 768 + 256 + c] = (_Float16)s1;
  S16[(size_t)n * 768 + 512 + c] = (_Float16)s2;
  if (c < UUQ) {
    float px = pos[n * 3], py = pos[n * 3 + 1], pz = pos[n * 3 + 2];
    float dot = px * DIRS[c * 3] + py * DIRS[c * 3 + 1] + pz * DIRS[c * 3 + 2];
#pragma unroll
    for (int kk = 0; kk < KKB; ++kk) {
      float ph = kappa[kk] * dot;
      cs16T[((size_t)(kk * 2 + 0) * UUQ + c) * NN + n] = (_Float16)cosf(ph);
      cs16T[((size_t)(kk * 2 + 1) * UUQ + c) * NN + n] = (_Float16)sinf(ph);
    }
  }
}

// ---------------- WMMA GEMMs ----------------
// QK = S16(N x 768) @ WqkT^T, elu+1 -> qf (row major), kfT (transposed)
__global__ void gemm_qk_kernel(const _Float16* __restrict__ S16, const _Float16* __restrict__ WqkT,
                               _Float16* __restrict__ qf, _Float16* __restrict__ kfT) {
  int tile = blockIdx.x * 4 + (threadIdx.x >> 5);  // 128*32
  int m0 = (tile >> 5) * 16, n0 = (tile & 31) * 16;
  const int l = threadIdx.x & 31;
  v8f acc = {};
  for (int kb = 0; kb < 768; kb += 32) {
    v16h a = frag_a_rowmajor(S16 + (size_t)m0 * 768 + kb, 768);
    v16h b = frag_b_kcontig(WqkT + (size_t)n0 * 768 + kb, 768);
    acc = WMMA_F16(a, b, acc);
  }
  int col = n0 + (l & 15);
#pragma unroll
  for (int v = 0; v < 8; ++v) {
    int row = m0 + ((l >> 4) * 8) + v;
    float x = acc[v];
    float e = (x > 0.f) ? (x + 1.f) : expf(x);  // elu(x)+1
    if (col < 256) qf[(size_t)row * 256 + col] = (_Float16)e;
    else           kfT[(size_t)(col - 256) * NN + row] = (_Float16)e;
  }
}

// V = feat16(N*9 x 256) @ WvT^T -> vbufT[h][c*9+l][n]
__global__ void gemm_v_kernel(const _Float16* __restrict__ feat16, const _Float16* __restrict__ WvT,
                              _Float16* __restrict__ vbufT) {
  int tile = blockIdx.x * 4 + (threadIdx.x >> 5);  // 1152*8
  int m0 = (tile >> 3) * 16, n0 = (tile & 7) * 16;
  const int l = threadIdx.x & 31;
  v8f acc = {};
  for (int kb = 0; kb < 256; kb += 32) {
    v16h a = frag_a_rowmajor(feat16 + (size_t)m0 * 256 + kb, 256);
    v16h b = frag_b_kcontig(WvT + (size_t)n0 * 256 + kb, 256);
    acc = WMMA_F16(a, b, acc);
  }
#pragma unroll
  for (int v = 0; v < 8; ++v) {
    int row = m0 + ((l >> 4) * 8) + v;  // n*9 + lq
    int col = n0 + (l & 15);            // h*16 + cv
    int n = row / 9, lq = row % 9, hh = col >> 4, cv = col & 15;
    vbufT[((size_t)hh * CLD + cv * 9 + lq) * NN + n] = (_Float16)acc[v];
  }
}

// Stage D: Mbuf[h][(trig,u,d)][cl] = sum_n cs[trig,u,n]*kf[d,n] * v[h][cl][n]
// block: 4 waves = 64 consecutive rows x one 16-col tile; B staged in LDS (TDM)
__global__ void gemm_fmm_M_kernel(const _Float16* __restrict__ cs16T, const _Float16* __restrict__ kfT,
                                  const _Float16* __restrict__ vbufT, float* __restrict__ Mbuf,
                                  int kk) {
  __shared__ __align__(16) _Float16 bt[16 * 1024];  // half-K staging, 32KB
  int blk = blockIdx.x;  // 8h * 25mg * 9nt
  int h   = blk / 225;
  int rem = blk % 225;
  int mg  = rem / 9;
  int n0  = (rem % 9) * 16;
  int wv  = threadIdx.x >> 5;
  int l   = threadIdx.x & 31;
  int m0  = mg * 64 + wv * 16;
  int trig = (m0 >= 800);
  int u    = (m0 >> 5) % 25;
  int d0   = m0 & 31;  // 0 or 16
  const _Float16* csrow = cs16T + (size_t)((kk * 2 + trig) * UUQ + u) * NN;
  const _Float16* kp    = kfT + (size_t)(h * 32 + d0 + (l & 15)) * NN;
  const _Float16* bsrc  = vbufT + (size_t)(h * CLD + n0) * NN;  // 16 rows x 2048 (contig rows)
  v8f acc = {};
  for (int ph = 0; ph < 2; ++ph) {
    __syncthreads();
#ifdef HAVE_TDM
    if (threadIdx.x < 32) {
      unsigned lds_off = (unsigned)(size_t)(&bt[0]);
      unsigned long long ga = (unsigned long long)(size_t)(bsrc + ph * 1024);
      u32x4 g0;
      g0[0] = 1u;                              // count=1
      g0[1] = lds_off;                         // lds_addr
      g0[2] = (unsigned)(ga & 0xffffffffu);    // global_addr lo
      g0[3] = (unsigned)(((ga >> 32) & 0x01ffffffu) | (2u << 30));  // addr hi | type=2
      i32x8 g1;
      g1[0] = 0x00010000;          // data_size = 2 bytes
      g1[1] = (int)(2048u << 16);  // tensor_dim0 = 2048
      g1[2] = (int)(144u << 16);   // tensor_dim1 = 144
      g1[3] = (int)(1024u << 16);  // tile_dim0 = 1024
      g1[4] = 16;                  // tile_dim1 = 16
      g1[5] = 2048;                // tensor_dim0_stride = 2048
      g1[6] = 0; g1[7] = 0;
      i32x4 z4; z4[0] = 0; z4[1] = 0; z4[2] = 0; z4[3] = 0;
      TDM_LOAD(g0, g1, z4);
      __builtin_amdgcn_s_wait_tensorcnt(0);
    }
    __syncthreads();
#else
    for (int i = threadIdx.x; i < 16 * 128; i += 128) {
      int rrow = i >> 7, c8 = i & 127;
      *(v8h*)(bt + rrow * 1024 + c8 * 8) =
          *(const v8h*)(bsrc + (size_t)rrow * NN + ph * 1024 + c8 * 8);
    }
    __syncthreads();
#endif
    const _Float16* btc = bt + (size_t)(l & 15) * 1024;
    for (int kb = 0; kb < 1024; kb += 32) {
      int o1 = ph * 1024 + kb + ((l >> 4) * 8);
      v8h k1 = *(const v8h*)(kp + o1);
      v8h k2 = *(const v8h*)(kp + o1 + 16);
      v8h c1 = *(const v8h*)(csrow + o1);
      v8h c2 = *(const v8h*)(csrow + o1 + 16);
      __builtin_prefetch(kp + o1 + 32, 0, 1);
      v16h a = pack_ab(c1 * k1, c2 * k2);
      const _Float16* bq = btc + kb + ((l >> 4) * 8);
      v16h b = pack_ab(*(const v8h*)(bq), *(const v8h*)(bq + 16));
      acc = WMMA_F16(a, b, acc);
    }
  }
#pragma unroll
  for (int v = 0; v < 8; ++v) {
    int row = m0 + ((l >> 4) * 8) + v;
    int col = n0 + (l & 15);
    Mbuf[((size_t)h * RWS + row) * CLD + col] = acc[v];
  }
}

// MB16T[h][c*9+m][r] = sum_l Mbuf[h][r][c*9+l] * B2[kk][u(r)][h][l][m]
__global__ void mb_kernel(const float* __restrict__ Mbuf, const float* __restrict__ B2,
                          _Float16* __restrict__ MB16T, int kk) {
  int idx = blockIdx.x * 256 + threadIdx.x;  // 8*1600*144
  int cm = idx % CLD;
  int r  = (idx / CLD) % RWS;
  int h  = idx / (CLD * RWS);
  int c = cm / 9, m = cm % 9;
  int u = ((r >= 800) ? (r - 800) : r) >> 5;
  const float* b2   = B2 + (((size_t)kk * UUQ + u) * HH + h) * 81;
  const float* mrow = Mbuf + ((size_t)h * RWS + r) * CLD + c * 9;
  float s = 0.f;
#pragma unroll
  for (int lq = 0; lq < 9; ++lq) s += mrow[lq] * b2[lq * 9 + m];
  MB16T[((size_t)h * CLD + cm) * RWS + r] = (_Float16)s;
}

// Stage E: out_pre[n][h][cm] += sum_g cs[.,n]*qf[n,.] * MB16T[h][cm][g]
// block: 4 waves = 64 consecutive n-rows x one 16-col cm tile; B (16x1600, contiguous) in LDS via TDM
__global__ void gemm_fmm_E_kernel(const _Float16* __restrict__ cs16T, const _Float16* __restrict__ qf,
                                  const _Float16* __restrict__ MB16T, float* __restrict__ out_pre,
                                  int kk) {
  __shared__ __align__(16) _Float16 bt[16 * RWS];  // 50KB
  int blk = blockIdx.x;  // 8h * 32ng * 9ct
  int h   = blk / 288;
  int rem = blk % 288;
  int ng  = rem / 9;
  int n0c = (rem % 9) * 16;
  int wv  = threadIdx.x >> 5;
  int l   = threadIdx.x & 31;
  int m0  = ng * 64 + wv * 16;
  const _Float16* bsrc = MB16T + (size_t)(h * CLD + n0c) * RWS;  // contiguous 16*1600 block
#ifdef HAVE_TDM
  if (threadIdx.x < 32) {
    unsigned lds_off = (unsigned)(size_t)(&bt[0]);
    unsigned long long ga = (unsigned long long)(size_t)bsrc;
    u32x4 g0;
    g0[0] = 1u;
    g0[1] = lds_off;
    g0[2] = (unsigned)(ga & 0xffffffffu);
    g0[3] = (unsigned)(((ga >> 32) & 0x01ffffffu) | (2u << 30));
    i32x8 g1;
    g1[0] = 0x00010000;           // data_size = 2 bytes
    g1[1] = (int)(25600u << 16);  // tensor_dim0 = 16*1600 (flat 1D)
    g1[2] = (int)(1u << 16);      // tensor_dim1 = 1
    g1[3] = (int)(25600u << 16);  // tile_dim0 = 25600
    g1[4] = 1;                    // tile_dim1 = 1
    g1[5] = 25600;                // tensor_dim0_stride
    g1[6] = 0; g1[7] = 0;
    i32x4 z4; z4[0] = 0; z4[1] = 0; z4[2] = 0; z4[3] = 0;
    TDM_LOAD(g0, g1, z4);
    __builtin_amdgcn_s_wait_tensorcnt(0);
  }
  __syncthreads();
#else
  for (int i = threadIdx.x; i < (16 * RWS) / 8; i += 128)
    ((v8h*)bt)[i] = ((const v8h*)bsrc)[i];
  __syncthreads();
#endif
  int nrow = m0 + (l & 15);
  const _Float16* qp = qf + (size_t)nrow * 256 + h * 32 + ((l >> 4) * 8);
  v8h q1 = *(const v8h*)(qp);
  v8h q2 = *(const v8h*)(qp + 16);
  v8f acc;
#pragma unroll
  for (int v = 0; v < 8; ++v) {
    int row = m0 + ((l >> 4) * 8) + v;
    acc[v] = out_pre[((size_t)row * HH + h) * CLD + n0c + (l & 15)];
  }
  const _Float16* btc = bt + (size_t)(l & 15) * RWS;
  const _Float16* csb = cs16T + (size_t)(kk * 2) * UUQ * NN + nrow;
  for (int kb = 0; kb < RWS; kb += 32) {
    int trig = (kb >= 800);
    int u = (kb >> 5) % 25;
    _Float16 ch = csb[(size_t)(trig * UUQ + u) * NN];
    v8h cv8;
#pragma unroll
    for (int i = 0; i < 8; ++i) cv8[i] = ch;
    v16h a = pack_ab(cv8 * q1, cv8 * q2);
    const _Float16* bq = btc + kb + ((l >> 4) * 8);
    v16h b = pack_ab(*(const v8h*)(bq), *(const v8h*)(bq + 16));
    acc = WMMA_F16(a, b, acc);
  }
#pragma unroll
  for (int v = 0; v < 8; ++v) {
    int row = m0 + ((l >> 4) * 8) + v;
    out_pre[((size_t)row * HH + h) * CLD + n0c + (l & 15)] = acc[v];
  }
}

__global__ void kfsum_kernel(const _Float16* __restrict__ kfT, float* __restrict__ kfsum) {
  int j = threadIdx.x;
  float s = 0.f;
  for (int n = 0; n < NN; ++n) s += (float)kfT[(size_t)j * NN + n];
  kfsum[j] = s;
}

__global__ void z_kernel(const _Float16* __restrict__ qf, const float* __restrict__ kfsum,
                         float* __restrict__ Z) {
  int idx = blockIdx.x * 256 + threadIdx.x;  // N*H
  if (idx >= NN * HH) return;
  int n = idx >> 3, h = idx & 7;
  float s = 0.f;
#pragma unroll
  for (int d = 0; d < 32; ++d) s += (float)qf[(size_t)n * 256 + h * 32 + d] * kfsum[h * 32 + d];
  Z[idx] = s + 1e-6f;
}

// aout[(n*9+mm)][hc] = out_pre[n][h][c*9+mm] / Z[n,h]   (row-major K-contig A for final GEMM)
__global__ void aout_kernel(const float* __restrict__ out_pre, const float* __restrict__ Z,
                            _Float16* __restrict__ aout) {
  int idx = blockIdx.x * 256 + threadIdx.x;  // 18432*128
  int hc = idx & 127;
  int g  = idx >> 7;
  int n = g / 9, mm = g % 9;
  int hh = hc >> 4, c = hc & 15;
  aout[idx] = (_Float16)(out_pre[((size_t)n * HH + hh) * CLD + c * 9 + mm] / Z[n * HH + hh]);
}

// out = aout(18432 x 128) @ WoT^T -> f32
__global__ void gemm_out_kernel(const _Float16* __restrict__ aout, const _Float16* __restrict__ WoT,
                                float* __restrict__ out) {
  int tile = blockIdx.x * 4 + (threadIdx.x >> 5);  // 1152*16
  int m0 = (tile >> 4) * 16, n0 = (tile & 15) * 16;
  const int l = threadIdx.x & 31;
  v8f acc = {};
  for (int kb = 0; kb < 128; kb += 32) {
    v16h a = frag_a_rowmajor(aout + (size_t)m0 * 128 + kb, 128);
    v16h b = frag_b_kcontig(WoT + (size_t)n0 * 128 + kb, 128);
    acc = WMMA_F16(a, b, acc);
  }
#pragma unroll
  for (int v = 0; v < 8; ++v) {
    int row = m0 + ((l >> 4) * 8) + v;
    int col = n0 + (l & 15);
    out[(size_t)row * 256 + col] = acc[v];
  }
}

// ================= host: constant tables (CG / Wigner / quadrature) ===============
static double factd(int n) { double r = 1.0; for (int i = 2; i <= n; ++i) r *= i; return r; }

static double cg_coeff(int j1, int m1, int j2, int m2, int j, int m) {
  if (m1 + m2 != m || j < std::abs(j1 - j2) || j > j1 + j2) return 0.0;
  double pref = std::sqrt((2.0 * j + 1.0) * factd(j + j1 - j2) * factd(j - j1 + j2) *
                          factd(j1 + j2 - j) / factd(j1 + j2 + j + 1));
  pref *= std::sqrt(factd(j + m) * factd(j - m) * factd(j1 - m1) * factd(j1 + m1) *
                    factd(j2 - m2) * factd(j2 + m2));
  double s = 0.0;
  for (int k = 0; k <= j1 + j2 - j; ++k) {
    int d0 = k, d1 = j1 + j2 - j - k, d2 = j1 - m1 - k, d3 = j2 + m2 - k,
        d4 = j - j2 + m1 + k, d5 = j - j1 - m2 + k;
    if (d0 < 0 || d1 < 0 || d2 < 0 || d3 < 0 || d4 < 0 || d5 < 0) continue;
    double t = 1.0 / (factd(d0) * factd(d1) * factd(d2) * factd(d3) * factd(d4) * factd(d5));
    s += (k & 1) ? -t : t;
  }
  return pref * s;
}

typedef std::complex<double> cplx;
static void umat(int l, cplx U[5][5]) {
  for (int i = 0; i < 5; ++i) for (int j = 0; j < 5; ++j) U[i][j] = 0.0;
  double s = 1.0 / std::sqrt(2.0);
  for (int m = -l; m <= l; ++m) {
    double sgn = (std::abs(m) & 1) ? -1.0 : 1.0;
    if (m > 0)      { U[l + m][l + m] = sgn * s; U[l + m][l - m] = s; }
    else if (m == 0){ U[l][l] = 1.0; }
    else            { U[l + m][l + m] = cplx(0.0, s); U[l + m][l - m] = cplx(0.0, -sgn * s); }
  }
}

static void real_cg_tab(int l1, int l2, int L, double out[5][5][5]) {
  int n1 = 2 * l1 + 1, n2 = 2 * l2 + 1, n3 = 2 * L + 1;
  double cgc[5][5][5];
  for (int i1 = 0; i1 < n1; ++i1)
    for (int i2 = 0; i2 < n2; ++i2)
      for (int i3 = 0; i3 < n3; ++i3)
        cgc[i1][i2][i3] = cg_coeff(l1, i1 - l1, l2, i2 - l2, L, i3 - L);
  cplx U1[5][5], U2[5][5], U3[5][5];
  umat(l1, U1); umat(l2, U2); umat(L, U3);
  cplx r[5][5][5];
  double nre = 0.0, nim = 0.0;
  for (int a = 0; a < n1; ++a)
    for (int b = 0; b < n2; ++b)
      for (int c = 0; c < n3; ++c) {
        cplx s = 0.0;
        for (int u = 0; u < n1; ++u)
          for (int v = 0; v < n2; ++v)
            for (int w = 0; w < n3; ++w)
              s += U1[a][u] * U2[b][v] * std::conj(U3[c][w]) * cgc[u][v][w];
        r[a][b][c] = s;
        nre += s.real() * s.real();
        nim += s.imag() * s.imag();
      }
  bool use_re = std::sqrt(nre) >= std::sqrt(nim);
  for (int a = 0; a < n1; ++a)
    for (int b = 0; b < n2; ++b)
      for (int c = 0; c < n3; ++c)
        out[a][b][c] = use_re ? r[a][b][c].real() : r[a][b][c].imag();
}

static void build_tables(float* YW, float* RRED, float* DIRS) {
  const int off[3] = {0, 1, 4};
  double Rd[9][9][9];
  std::memset(Rd, 0, sizeof(Rd));
  double cnt[3] = {0, 0, 0};
  for (int l1 = 0; l1 <= 2; ++l1)
    for (int l2 = 0; l2 <= 2; ++l2)
      for (int L = std::abs(l1 - l2); L <= std::min(l1 + l2, 2); ++L) {
        double rc[5][5][5];
        real_cg_tab(l1, l2, L, rc);
        for (int i1 = 0; i1 < 2 * l1 + 1; ++i1)
          for (int i2 = 0; i2 < 2 * l2 + 1; ++i2)
            for (int i3 = 0; i3 < 2 * L + 1; ++i3)
              Rd[off[l1] + i1][off[l2] + i2][off[L] + i3] += rc[i1][i2][i3];
        cnt[L] += 1.0;
      }
  for (int L = 0; L <= 2; ++L) {
    double dn = std::max(cnt[L], 1.0);
    for (int a = 0; a < 9; ++a)
      for (int b = 0; b < 9; ++b)
        for (int c = 0; c < 2 * L + 1; ++c)
          Rd[a][b][off[L] + c] /= dn;
  }
  for (int a = 0; a < 9; ++a)
    for (int b = 0; b < 9; ++b)
      for (int c = 0; c < 9; ++c)
        RRED[(a * 9 + b) * 9 + c] = (float)Rd[a][b][c];
  const double PI = 3.14159265358979323846;
  const double xg[5] = {-0.906179845938664, -0.5384693101056831, 0.0,
                        0.5384693101056831, 0.906179845938664};
  const double wg[5] = {0.23692688505618908, 0.47862867049936647, 0.5688888888888889,
                        0.47862867049936647, 0.23692688505618908};
  for (int i = 0; i < 5; ++i)
    for (int j = 0; j < 5; ++j) {
      int u = i * 5 + j;
      double ct = xg[i], w = wg[i] * (2.0 * PI / 5.0);
      double ph = 2.0 * PI * j / 5.0;
      double st = std::sqrt(std::max(1.0 - ct * ct, 0.0));
      double x = st * std::cos(ph), y = st * std::sin(ph), z = ct;
      DIRS[u * 3 + 0] = (float)x;
      DIRS[u * 3 + 1] = (float)y;
      DIRS[u * 3 + 2] = (float)z;
      double Y[9] = {0.282095, 0.488603 * y, 0.488603 * z, 0.488603 * x,
                     1.092548 * x * y, 1.092548 * y * z, 0.315392 * (3 * z * z - 1),
                     1.092548 * x * z, 0.546274 * (x * x - y * y)};
      for (int e = 0; e < 9; ++e) YW[e * 25 + u] = (float)(Y[e] * w);
    }
}

// ================= launch =================
extern "C" void kernel_launch(void* const* d_in, const int* in_sizes, int n_in,
                              void* d_out, int out_size, void* d_ws, size_t ws_size,
                              hipStream_t stream) {
  const float* node_pos  = (const float*)d_in[0];
  const float* node_feat = (const float*)d_in[1];
  const float* Wq        = (const float*)d_in[2];
  const float* Wk        = (const float*)d_in[3];
  const float* Wv        = (const float*)d_in[4];
  const float* Wo        = (const float*)d_in[5];
  const float* a_in      = (const float*)d_in[6];
  const float* kappa     = (const float*)d_in[7];

  static float h_tab[1029];  // YW(225) | RRED(729) | DIRS(75)
  build_tables(h_tab, h_tab + 225, h_tab + 954);

  char* ws = (char*)d_ws;
  size_t cur = 0;
  auto alloc = [&](size_t bytes) { size_t o = cur; cur += (bytes + 255) & ~(size_t)255; return o; };

  size_t o_tab    = alloc(1029 * 4);
  size_t o_b2     = alloc((size_t)KKB * UUQ * HH * 81 * 4);
  size_t o_s16    = alloc((size_t)NN * 768 * 2);
  size_t o_wqkT   = alloc((size_t)512 * 768 * 2);
  size_t o_qf     = alloc((size_t)NN * 256 * 2);
  size_t o_kfT    = alloc((size_t)256 * NN * 2);
  size_t o_kfsum  = alloc(256 * 4);
  size_t o_feat16 = alloc((size_t)NN * 9 * 256 * 2);
  size_t o_wvT    = alloc((size_t)128 * 256 * 2);
  size_t o_woT    = alloc((size_t)256 * 128 * 2);
  size_t o_vbufT  = alloc((size_t)HH * CLD * NN * 2);
  size_t o_cs16   = alloc((size_t)KKB * 2 * UUQ * NN * 2);
  size_t o_mbuf   = alloc((size_t)HH * RWS * CLD * 4);
  size_t o_mb16T  = alloc((size_t)HH * CLD * RWS * 2);
  size_t o_opre   = alloc((size_t)NN * HH * CLD * 4);
  size_t o_z      = alloc((size_t)NN * HH * 4);
  size_t o_aout   = alloc((size_t)18432 * 128 * 2);
  (void)ws_size;

  const float* d_yw   = (const float*)(ws + o_tab);
  const float* d_rred = d_yw + 225;
  const float* d_dirs = d_yw + 954;
  float*     d_b2     = (float*)(ws + o_b2);
  _Float16*  d_s16    = (_Float16*)(ws + o_s16);
  _Float16*  d_wqkT   = (_Float16*)(ws + o_wqkT);
  _Float16*  d_qf     = (_Float16*)(ws + o_qf);
  _Float16*  d_kfT    = (_Float16*)(ws + o_kfT);
  float*     d_kfsum  = (float*)(ws + o_kfsum);
  _Float16*  d_feat16 = (_Float16*)(ws + o_feat16);
  _Float16*  d_wvT    = (_Float16*)(ws + o_wvT);
  _Float16*  d_woT    = (_Float16*)(ws + o_woT);
  _Float16*  d_vbufT  = (_Float16*)(ws + o_vbufT);
  _Float16*  d_cs16   = (_Float16*)(ws + o_cs16);
  float*     d_mbuf   = (float*)(ws + o_mbuf);
  _Float16*  d_mb16T  = (_Float16*)(ws + o_mb16T);
  float*     d_opre   = (float*)(ws + o_opre);
  float*     d_z      = (float*)(ws + o_z);
  _Float16*  d_aout   = (_Float16*)(ws + o_aout);

  hipMemcpyAsync(ws + o_tab, h_tab, 1029 * 4, hipMemcpyHostToDevice, stream);
  hipMemsetAsync(d_opre, 0, (size_t)NN * HH * CLD * 4, stream);

  pack_wqkT_kernel<<<(512 * 768) / 256, 256, 0, stream>>>(Wq, Wk, d_wqkT);
  t16_kernel<<<(256 * 128 + 255) / 256, 256, 0, stream>>>(Wv, d_wvT, 256, 128);
  t16_kernel<<<(128 * 256 + 255) / 256, 256, 0, stream>>>(Wo, d_woT, 128, 256);
  b2_kernel<<<(KKB * UUQ * HH * 81 + 255) / 256, 256, 0, stream>>>(d_yw, d_rred, a_in, d_b2);
  prep_kernel<<<NN, 256, 0, stream>>>(node_pos, node_feat, kappa, d_dirs, d_s16, d_feat16, d_cs16);

  gemm_qk_kernel<<<(128 * 32) / 4, 128, 0, stream>>>(d_s16, d_wqkT, d_qf, d_kfT);
  kfsum_kernel<<<1, 256, 0, stream>>>(d_kfT, d_kfsum);
  gemm_v_kernel<<<(1152 * 8) / 4, 128, 0, stream>>>(d_feat16, d_wvT, d_vbufT);

  for (int kk = 0; kk < KKB; ++kk) {
    gemm_fmm_M_kernel<<<(HH * 25 * 9), 128, 0, stream>>>(d_cs16, d_kfT, d_vbufT, d_mbuf, kk);
    mb_kernel<<<(HH * RWS * CLD) / 256, 256, 0, stream>>>(d_mbuf, d_b2, d_mb16T, kk);
    gemm_fmm_E_kernel<<<(HH * 32 * 9), 128, 0, stream>>>(d_cs16, d_qf, d_mb16T, d_opre, kk);
  }

  z_kernel<<<(NN * HH + 255) / 256, 256, 0, stream>>>(d_qf, d_kfsum, d_z);
  aout_kernel<<<(18432 * 128) / 256, 256, 0, stream>>>(d_opre, d_z, d_aout);
  gemm_out_kernel<<<(1152 * 16) / 4, 128, 0, stream>>>(d_aout, d_woT, (float*)d_out);
}
